// QwenVisionAttention_84696755077355
// MI455X (gfx1250) — matrix operand order repaired and checked by reading, not compiled, not run on hardware
//
#include <hip/hip_runtime.h>
#include <hip/hip_bf16.h>

typedef _Float16 v16h __attribute__((ext_vector_type(16)));
typedef _Float16 v8h  __attribute__((ext_vector_type(8)));
typedef float    v8f  __attribute__((ext_vector_type(8)));

union H16 { v16h v; v8h h[2]; };

#define WMMA_F16(a, b, c) \
  __builtin_amdgcn_wmma_f32_16x16x32_f16(false, (a), false, (b), (short)0, (c), false, false)

// ---------------- problem constants ----------------
constexpr int Bb   = 2;
constexpr int Nn   = 2048;
constexpr int Cc   = 1664;
constexpr int Hh   = 16;
constexpr int Dd   = 104;   // head dim
constexpr int Dp   = 128;   // padded head dim
constexpr int C3   = 3 * Cc;
constexpr float SCALE = 0.09805806756909201f;   // 1/sqrt(104)
constexpr float LOG2E = 1.4426950408889634f;

// ---- CDNA5 async global->LDS copy (ASYNCcnt-tracked DMA), inline asm path ----
__device__ __forceinline__ void async_b128(const void* gsrc, void* ldst) {
  // low 32 bits of a flat LDS pointer are the DS byte offset (ISA 10.2 aperture layout)
  unsigned loff = (unsigned)(unsigned long long)ldst;
  unsigned long long ga = (unsigned long long)gsrc;
  asm volatile("global_load_async_to_lds_b128 %0, %1, off"
               :: "v"(loff), "v"(ga) : "memory");
}
__device__ __forceinline__ void wait_async0() {
  asm volatile("s_wait_asynccnt 0x0" ::: "memory");
}

// ---------------- fp32 -> fp16 convert ----------------
__global__ void cvt_f32_f16(const float* __restrict__ in, _Float16* __restrict__ out, int n) {
  int i = blockIdx.x * blockDim.x + threadIdx.x;
  if (i < n) out[i] = (_Float16)in[i];
}

// ---------------- WMMA GEMM: f16 A (MxK row-major) * f16 B (KxN row-major) ----------------
// one wave computes a 32x64 output tile; K stepped by 32; B operands reused by 2 A rows-tiles.
// A-operand (16x32 f16): lane<16 -> row=lane, k {0..7,16..23}; lane>=16 -> row=lane-16, k {8..15,24..31}
// B-operand (32x16 f16): col = lane&15, 16 K values per lane in the same k order as A.
__global__ void gemm_f16out(const _Float16* __restrict__ A, const _Float16* __restrict__ Bw,
                            const float* __restrict__ bias, _Float16* __restrict__ Cout,
                            int K, int Nld) {
  const int lane = threadIdx.x & 31;
  const int m0 = blockIdx.x * 32;
  const int n0 = blockIdx.y * 64;
  const int row  = lane & 15;
  const int koff = (lane & 16) ? 8 : 0;

  v8f acc[2][4] = {};

  for (int k = 0; k < K; k += 32) {
    const _Float16* ap0 = A + (size_t)(m0 + row) * K + k + koff;
    const _Float16* ap1 = ap0 + (size_t)16 * K;
    H16 Av0; Av0.h[0] = *(const v8h*)(ap0); Av0.h[1] = *(const v8h*)(ap0 + 16);
    H16 Av1; Av1.h[0] = *(const v8h*)(ap1); Av1.h[1] = *(const v8h*)(ap1 + 16);
    const _Float16* bp = Bw + (size_t)(k + koff) * Nld + n0 + row;
    if (k + 32 < K) __builtin_prefetch(bp + (size_t)32 * Nld, 0, 0);
    v16h B0, B1, B2, B3;
#pragma unroll
    for (int s = 0; s < 16; ++s) {
      const size_t o = (size_t)(s < 8 ? s : s + 8) * Nld;
      B0[s] = bp[o]; B1[s] = bp[o + 16]; B2[s] = bp[o + 32]; B3[s] = bp[o + 48];
    }
    acc[0][0] = WMMA_F16(Av0.v, B0, acc[0][0]);
    acc[1][0] = WMMA_F16(Av1.v, B0, acc[1][0]);
    acc[0][1] = WMMA_F16(Av0.v, B1, acc[0][1]);
    acc[1][1] = WMMA_F16(Av1.v, B1, acc[1][1]);
    acc[0][2] = WMMA_F16(Av0.v, B2, acc[0][2]);
    acc[1][2] = WMMA_F16(Av1.v, B2, acc[1][2]);
    acc[0][3] = WMMA_F16(Av0.v, B3, acc[0][3]);
    acc[1][3] = WMMA_F16(Av1.v, B3, acc[1][3]);
  }

  const int hh = lane >> 4, col0 = lane & 15;
#pragma unroll
  for (int i = 0; i < 2; ++i) {
#pragma unroll
    for (int j = 0; j < 4; ++j) {
      const int col = n0 + 16 * j + col0;
      const float bv = bias[col];
#pragma unroll
      for (int r = 0; r < 8; ++r) {
        const int rr = m0 + 16 * i + r + 8 * hh;
        Cout[(size_t)rr * Nld + col] = (_Float16)(acc[i][j][r] + bv);
      }
    }
  }
}

// same, fp32 output (final projection)
__global__ void gemm_f32out(const _Float16* __restrict__ A, const _Float16* __restrict__ Bw,
                            const float* __restrict__ bias, float* __restrict__ Cout,
                            int K, int Nld) {
  const int lane = threadIdx.x & 31;
  const int m0 = blockIdx.x * 32;
  const int n0 = blockIdx.y * 64;
  const int row  = lane & 15;
  const int koff = (lane & 16) ? 8 : 0;

  v8f acc[2][4] = {};

  for (int k = 0; k < K; k += 32) {
    const _Float16* ap0 = A + (size_t)(m0 + row) * K + k + koff;
    const _Float16* ap1 = ap0 + (size_t)16 * K;
    H16 Av0; Av0.h[0] = *(const v8h*)(ap0); Av0.h[1] = *(const v8h*)(ap0 + 16);
    H16 Av1; Av1.h[0] = *(const v8h*)(ap1); Av1.h[1] = *(const v8h*)(ap1 + 16);
    const _Float16* bp = Bw + (size_t)(k + koff) * Nld + n0 + row;
    if (k + 32 < K) __builtin_prefetch(bp + (size_t)32 * Nld, 0, 0);
    v16h B0, B1, B2, B3;
#pragma unroll
    for (int s = 0; s < 16; ++s) {
      const size_t o = (size_t)(s < 8 ? s : s + 8) * Nld;
      B0[s] = bp[o]; B1[s] = bp[o + 16]; B2[s] = bp[o + 32]; B3[s] = bp[o + 48];
    }
    acc[0][0] = WMMA_F16(Av0.v, B0, acc[0][0]);
    acc[1][0] = WMMA_F16(Av1.v, B0, acc[1][0]);
    acc[0][1] = WMMA_F16(Av0.v, B1, acc[0][1]);
    acc[1][1] = WMMA_F16(Av1.v, B1, acc[1][1]);
    acc[0][2] = WMMA_F16(Av0.v, B2, acc[0][2]);
    acc[1][2] = WMMA_F16(Av1.v, B2, acc[1][2]);
    acc[0][3] = WMMA_F16(Av0.v, B3, acc[0][3]);
    acc[1][3] = WMMA_F16(Av1.v, B3, acc[1][3]);
  }

  const int hh = lane >> 4, col0 = lane & 15;
#pragma unroll
  for (int i = 0; i < 2; ++i) {
#pragma unroll
    for (int j = 0; j < 4; ++j) {
      const int col = n0 + 16 * j + col0;
      const float bv = bias[col];
#pragma unroll
      for (int r = 0; r < 8; ++r) {
        const int rr = m0 + 16 * i + r + 8 * hh;
        Cout[(size_t)rr * Nld + col] = acc[i][j][r] + bv;
      }
    }
  }
}

// ---------------- RoPE + scatter to Q / K^T / V (f16, head-dim padded to 128) ----------------
__global__ void rope_scatter(const _Float16* __restrict__ QKV, const int* __restrict__ coords,
                             _Float16* __restrict__ Qr, _Float16* __restrict__ Kr,
                             _Float16* __restrict__ Vw) {
  const int t = blockIdx.x * blockDim.x + threadIdx.x;
  const int d = t & 127;
  const int h = (t >> 7) & 15;
  const int n = (t >> 11) & 2047;
  const int b = t >> 22;
  if (b >= Bb) return;

  const size_t q_off = ((size_t)((b * Hh + h) * Nn) + n) * Dp + d;   // [b][h][n][128]
  const size_t k_off = ((size_t)((b * Hh + h) * Dp) + d) * Nn + n;   // [b][h][128][n] (transposed)
  const size_t v_off = q_off;

  if (d >= Dd) {
    Qr[q_off] = (_Float16)0.0f;
    Kr[k_off] = (_Float16)0.0f;
    Vw[v_off] = (_Float16)0.0f;
    return;
  }

  int part, o, dp_;
  if (d < 34)      { part = 0; o = d;      dp_ = 34; }
  else if (d < 68) { part = 1; o = d - 34; dp_ = 34; }
  else             { part = 2; o = d - 68; dp_ = 36; }
  const int halfp = dp_ >> 1;
  const int j = (o < halfp) ? o : (o - halfp);
  const float coord = (float)coords[((size_t)(b * Nn + n)) * 3 + part];
  const float ang = coord * __expf(-(2.0f * (float)j / (float)dp_) * 9.210340371976184f);
  const float cs = __cosf(ang), sn = __sinf(ang);

  const int dpart = (d < 52) ? d + 52 : d - 52;
  const float sgn = (d < 52) ? -1.0f : 1.0f;

  const size_t rowbase = (size_t)(b * Nn + n) * C3;
  const size_t hd  = (size_t)h * Dd;
  const float qv  = (float)QKV[rowbase + hd + d];
  const float qvp = (float)QKV[rowbase + hd + dpart];
  const float kv  = (float)QKV[rowbase + Cc + hd + d];
  const float kvp = (float)QKV[rowbase + Cc + hd + dpart];
  const float vv  = (float)QKV[rowbase + 2 * Cc + hd + d];

  Qr[q_off] = (_Float16)(qv * cs + sgn * qvp * sn);
  Kr[k_off] = (_Float16)(kv * cs + sgn * kvp * sn);
  Vw[v_off] = (_Float16)vv;
}

// ---------------- flash attention ----------------
// block = 8 waves, all on the same (b,h): 8 consecutive 16-query tiles.
// K/V chunks (32 keys) staged into LDS by cooperative async DMA, double-buffered.
__global__ __launch_bounds__(256) void attn_kernel(const _Float16* __restrict__ Qr,
                                                   const _Float16* __restrict__ Kr,
                                                   const _Float16* __restrict__ Vw,
                                                   _Float16* __restrict__ O16) {
  __shared__ __align__(64) _Float16 ldsK[2][128 * 32];  // [d][tok]   8KB x2
  __shared__ __align__(64) _Float16 ldsV[2][32 * 128];  // [tok][d]   8KB x2
  __shared__ __align__(64) _Float16 ldsP[8][16 * 32];   // per-wave P tile, 1KB x8

  const int tid  = threadIdx.x;
  const int lane = tid & 31;
  const int wid  = tid >> 5;
  const int bh   = blockIdx.x >> 4;          // b*16+h (0..31)
  const int qg   = blockIdx.x & 15;          // query-tile group within (b,h)
  const int nt   = qg * 8 + wid;             // this wave's 16-query tile (0..127)
  const int b    = bh >> 4;
  const int h    = bh & 15;

  const int row  = lane & 15;
  const int hh   = lane >> 4;
  const int col0 = lane & 15;
  const int koff = (lane & 16) ? 8 : 0;

  _Float16* ldsW = &ldsP[wid][0];

  // preload Q tile in A-operand layout
  const _Float16* qbase = Qr + ((size_t)(bh * Nn) + nt * 16) * Dp;
  v16h Qa[4];
#pragma unroll
  for (int kc = 0; kc < 4; ++kc) {
    const _Float16* qp = qbase + row * Dp + kc * 32 + koff;
    H16 t; t.h[0] = *(const v8h*)(qp); t.h[1] = *(const v8h*)(qp + 16);
    Qa[kc] = t.v;
  }

  const _Float16* kbase = Kr + (size_t)bh * Dp * Nn;   // [d][token]
  const _Float16* vbase = Vw + (size_t)bh * Nn * Dp;   // [token][d]

  float m[8], l[8];
  v8f acc[8] = {};
#pragma unroll
  for (int r = 0; r < 8; ++r) { m[r] = -1.0e30f; l[r] = 0.0f; }

  // stage chunk 0 into buffer 0 (4 async b128 per thread per chunk: 16KB / 256 threads)
  {
#pragma unroll
    for (int rr = 0; rr < 2; ++rr) {
      const int p = tid + 256 * rr;
      const int d = p >> 2, seg = p & 3;                    // K tile: 128 rows x 4 segs
      async_b128(kbase + (size_t)d * Nn + seg * 8, &ldsK[0][d * 32 + seg * 8]);
      const int tok = p >> 4, sg = p & 15;                  // V tile: 32 rows x 16 segs
      async_b128(vbase + (size_t)tok * Dp + sg * 8, &ldsV[0][tok * 128 + sg * 8]);
    }
  }

  for (int t0 = 0, it = 0; t0 < Nn; t0 += 32, ++it) {
    const int cb = it & 1;
    wait_async0();        // our share of current buffer landed
    __syncthreads();      // everyone's share landed; all reads of the other buffer done

    if (t0 + 32 < Nn) {   // prefetch next chunk into the other buffer
      const int nb = cb ^ 1;
#pragma unroll
      for (int rr = 0; rr < 2; ++rr) {
        const int p = tid + 256 * rr;
        const int d = p >> 2, seg = p & 3;
        async_b128(kbase + (size_t)d * Nn + (t0 + 32) + seg * 8, &ldsK[nb][d * 32 + seg * 8]);
        const int tok = p >> 4, sg = p & 15;
        async_b128(vbase + (size_t)(t0 + 32 + tok) * Dp + sg * 8, &ldsV[nb][tok * 128 + sg * 8]);
      }
    }

    // ---- S = Q K^T for 32 keys (two 16x16 tiles), K operand from LDS ----
    v8f clo = {}, chi = {};
#pragma unroll
    for (int kc = 0; kc < 4; ++kc) {
      const _Float16* kp = &ldsK[cb][(kc * 32 + koff) * 32 + col0];
      v16h Blo, Bhi;
#pragma unroll
      for (int e = 0; e < 16; ++e) {
        const int o = (e < 8 ? e : e + 8) * 32;
        Blo[e] = kp[o]; Bhi[e] = kp[o + 16];
      }
      clo = WMMA_F16(Qa[kc], Blo, clo);
      chi = WMMA_F16(Qa[kc], Bhi, chi);
    }

    // ---- online softmax (rows live in 16-lane halves) ----
    float alpha[8];
#pragma unroll
    for (int r = 0; r < 8; ++r) {
      float slo = clo[r] * SCALE;
      float shi = chi[r] * SCALE;
      float mx = fmaxf(slo, shi);
      mx = fmaxf(mx, __shfl_xor(mx, 1));
      mx = fmaxf(mx, __shfl_xor(mx, 2));
      mx = fmaxf(mx, __shfl_xor(mx, 4));
      mx = fmaxf(mx, __shfl_xor(mx, 8));
      const float mn = fmaxf(m[r], mx);
      const float a  = exp2f((m[r] - mn) * LOG2E);
      m[r] = mn;
      const float pl = exp2f((slo - mn) * LOG2E);
      const float ph = exp2f((shi - mn) * LOG2E);
      float rs = pl + ph;
      rs += __shfl_xor(rs, 1);
      rs += __shfl_xor(rs, 2);
      rs += __shfl_xor(rs, 4);
      rs += __shfl_xor(rs, 8);
      l[r] = l[r] * a + rs;
      alpha[r] = a;
      const int Mr = r + 8 * hh;
      ldsW[Mr * 32 + col0]      = (_Float16)pl;
      ldsW[Mr * 32 + 16 + col0] = (_Float16)ph;
    }
#pragma unroll
    for (int s = 0; s < 8; ++s)
#pragma unroll
      for (int r = 0; r < 8; ++r) acc[s][r] *= alpha[r];

    // P tile: C-layout -> A-layout transpose via own LDS tile (same-wave DS is in-order)
    const _Float16* pp = ldsW + row * 32 + koff;
    H16 Pt; Pt.h[0] = *(const v8h*)(pp); Pt.h[1] = *(const v8h*)(pp + 16);

    // ---- acc += P * V, V operand from LDS ----
    const _Float16* vp = &ldsV[cb][koff * 128 + col0];
#pragma unroll
    for (int s = 0; s < 8; ++s) {
      v16h Bv;
#pragma unroll
      for (int e = 0; e < 16; ++e) Bv[e] = vp[(e < 8 ? e : e + 8) * 128 + 16 * s];
      acc[s] = WMMA_F16(Pt.v, Bv, acc[s]);
    }
  }

  // ---- normalize and store out[b][token][h][d] (f16, d < 104) ----
  float inv[8];
#pragma unroll
  for (int r = 0; r < 8; ++r) inv[r] = 1.0f / l[r];
#pragma unroll
  for (int s = 0; s < 8; ++s) {
    const int d = 16 * s + col0;
    if (d < Dd) {
#pragma unroll
      for (int r = 0; r < 8; ++r) {
        const int tok = nt * 16 + r + 8 * hh;
        O16[((size_t)(b * Nn + tok) * Hh + h) * Dd + d] = (_Float16)(acc[s][r] * inv[r]);
      }
    }
  }
}

// ---------------- host launcher ----------------
extern "C" void kernel_launch(void* const* d_in, const int* in_sizes, int n_in,
                              void* d_out, int out_size, void* d_ws, size_t ws_size,
                              hipStream_t stream) {
  (void)in_sizes; (void)n_in; (void)out_size; (void)ws_size;

  const float* x      = (const float*)d_in[0];
  const int*   coords = (const int*)d_in[1];
  const float* qkv_w  = (const float*)d_in[2];
  const float* qkv_b  = (const float*)d_in[3];
  const float* proj_w = (const float*)d_in[4];
  const float* proj_b = (const float*)d_in[5];
  float* out = (float*)d_out;

  char* ws = (char*)d_ws;
  _Float16* X16   = (_Float16*)(ws);                  // 4096x1664          13,631,488 B
  _Float16* Wqkv  = (_Float16*)(ws + 13631488ull);    // 1664x4992          16,613,376 B
  _Float16* QKV16 = (_Float16*)(ws + 30244864ull);    // 4096x4992          40,894,464 B
  _Float16* Qr    = (_Float16*)(ws + 71139328ull);    // [b][h][2048][128]  16,777,216 B
  _Float16* Kr    = (_Float16*)(ws + 87916544ull);    // [b][h][128][2048]  16,777,216 B
  _Float16* Vw    = (_Float16*)(ws + 104693760ull);   // [b][h][2048][128]  16,777,216 B
  _Float16* O16   = (_Float16*)(ws + 121470976ull);   // 4096x1664          13,631,488 B
  _Float16* Wproj = (_Float16*)(ws + 135102464ull);   // 1664x1664           5,537,792 B

  // 1) fp32 -> fp16 conversions
  {
    int n = Bb * Nn * Cc;
    cvt_f32_f16<<<(n + 255) / 256, 256, 0, stream>>>(x, X16, n);
    n = Cc * C3;
    cvt_f32_f16<<<(n + 255) / 256, 256, 0, stream>>>(qkv_w, Wqkv, n);
    n = Cc * Cc;
    cvt_f32_f16<<<(n + 255) / 256, 256, 0, stream>>>(proj_w, Wproj, n);
  }

  // 2) qkv = x @ qkv_w + qkv_b   (4096 x 4992, K = 1664), 32x64 tile per wave
  gemm_f16out<<<dim3(Bb * Nn / 32, C3 / 64), 32, 0, stream>>>(X16, Wqkv, qkv_b, QKV16, Cc, C3);

  // 3) RoPE + scatter into Q / K^T / V, padded head dim 128
  {
    int n = Bb * Nn * Hh * Dp;
    rope_scatter<<<n / 256, 256, 0, stream>>>(QKV16, coords, Qr, Kr, Vw);
  }

  // 4) flash attention -> O16 [b][n][h*104]; 16 blocks per (b,h), 8 waves each
  attn_kernel<<<Bb * Hh * 16, 256, 0, stream>>>(Qr, Kr, Vw, O16);

  // 5) out = O @ proj_w + proj_b  (4096 x 1664, K = 1664), fp32 output
  gemm_f32out<<<dim3(Bb * Nn / 32, Cc / 64), 32, 0, stream>>>(O16, Wproj, proj_b, out, Cc, Cc);
}
